// PixelBlock_83958020703015
// MI455X (gfx1250) — compile-verified, matrix operand order checked by reading, not run on hardware
//
#include <hip/hip_runtime.h>
#include <hip/hip_bf16.h>

// ---------------------------------------------------------------------------
// PixelBlock forward for MI455X (gfx1250, wave32, WMMA).
// All GEMM-shaped work (causal convs, pointwise convs, q/k/v, QK^T, PV,
// output proj) runs on v_wmma_f32_16x16x32_f16 with f32 accumulation.
// Causal conv padding is a materialized zero halo (no exec divergence in the
// inner loops); conv and pointwise GEMMs are register-tiled 2x2; all hot-path
// indexing is 32-bit unsigned so addresses fold into saddr+voffset form.
// ---------------------------------------------------------------------------

typedef __attribute__((ext_vector_type(16))) _Float16 v16h;
typedef __attribute__((ext_vector_type(8)))  float    v8f;

#define B_    8
#define C_    256u
#define H_    32u
#define W_    32u
#define S_    1024u         // H*W
#define PH_   33u           // H + 1 (top halo row)
#define PW_   34u           // W + 2 (left halo cols)
#define HEADS 8u
#define NEGINF (-3.0e38f)

union V16 { v16h v; uint4 q[2]; _Float16 h[16]; };

__device__ __forceinline__ float eluf(float x) { return x > 0.f ? x : (__expf(x) - 1.f); }
__device__ __forceinline__ float sigm(float x) { return 1.f / (1.f + __expf(-x)); }

__device__ __forceinline__ v8f wmma32(const V16& a, const V16& b, v8f c) {
  return __builtin_amdgcn_wmma_f32_16x16x32_f16(false, a.v, false, b.v,
                                                (short)0, c, false, false);
}
__device__ __forceinline__ uint4 ld4(const _Float16* p) { return *(const uint4*)p; }

// WMMA f16 operand index maps (wave32):
//  A (16xK) / B (Kx16, symmetric): lane = idx16 + (khi<<4), K = (e&7)+(e&8?16:0)+(lane>=16?8:0)
//  C/D (16x16 f32): lane holds N=lane&15, VGPR r holds M = r + (lane>=16?8:0)

// --- zero-fill (uint4 granularity) -----------------------------------------
__global__ void k_fill0(uint4* __restrict__ p, unsigned n4) {
  unsigned i = blockIdx.x * blockDim.x + threadIdx.x;
  if (i >= n4) return;
  uint4 z; z.x = z.y = z.z = z.w = 0u;
  p[i] = z;
}

// --- NCHW f32 -> padded NHWC f16 elu(x), zero halo --------------------------
__global__ void k_prep_elu_pad(const float* __restrict__ x, _Float16* __restrict__ e) {
  unsigned i = blockIdx.x * blockDim.x + threadIdx.x;  // over B*PH*PW*C
  if (i >= B_ * PH_ * PW_ * C_) return;
  unsigned c  = i & (C_ - 1u);
  unsigned p  = i >> 8;
  unsigned pw = p % PW_;
  unsigned ph = (p / PW_) % PH_;
  unsigned b  = p / (PW_ * PH_);
  float v = 0.f;
  if (ph >= 1u && pw >= 2u)
    v = eluf(x[(b * C_ + c) * S_ + (ph - 1u) * W_ + (pw - 2u)]);
  e[i] = (_Float16)v;
}

// --- pack f32 weight matrix into per-lane WMMA B-operand layout -------------
// dst[((kt*NT + nt)*32 + lane)*16 + e] = src[n*strideN + k*strideK + off]
__global__ void k_pack(const float* __restrict__ src, _Float16* __restrict__ dst,
                       unsigned KT, unsigned NT, unsigned strideN, unsigned strideK,
                       unsigned off) {
  unsigned t = blockIdx.x * blockDim.x + threadIdx.x;
  if (t >= KT * NT * 32u) return;
  unsigned lane = t & 31u;
  unsigned kg   = (lane >> 4) * 8u;
  unsigned nt   = (t >> 5) % NT;
  unsigned kt   = t / (32u * NT);
  unsigned n    = nt * 16u + (lane & 15u);
  alignas(16) _Float16 tmp[16];
#pragma unroll
  for (unsigned e = 0; e < 16u; ++e) {
    unsigned k = kt * 32u + (e & 7u) + ((e & 8u) ? 16u : 0u) + kg;
    tmp[e] = (_Float16)src[n * strideN + k * strideK + off];
  }
  _Float16* d = dst + t * 16u;
  *(uint4*)d       = *(uint4*)tmp;
  *(uint4*)(d + 8) = *(uint4*)(tmp + 8);
}

// --- causal 2x3 conv as 6-tap shifted WMMA GEMM, 2x2 register tiling --------
// wave computes 32 pixels (full W row) x 32 out-channels.
// mode 0: out16(padded NHWC) = f16(elu(acc+bias))        (conv1, NP=8)
// mode 1: out32(plain NHWC [B,S,Cout]) = acc+bias        (conv2, NP=16)
__global__ void __launch_bounds__(256)
k_conv(const _Float16* __restrict__ Ain,   // padded [B][PH][PW][C]
       const _Float16* __restrict__ Bp, const float* __restrict__ bias,
       unsigned NP, _Float16* __restrict__ out16, float* __restrict__ out32, int mode) {
  unsigned wave = threadIdx.x >> 5;
  unsigned lane = threadIdx.x & 31u;
  unsigned tile = blockIdx.x * (blockDim.x >> 5) + wave;
  unsigned mt = tile / NP;                 // 0..255 : (b, h)
  unsigned np = tile % NP;
  unsigned h  = mt & 31u;
  unsigned b  = mt >> 5;
  unsigned m  = lane & 15u;
  unsigned kg = (lane >> 4) * 8u;
  unsigned NT = NP * 2u;

  v8f a00 = {}, a01 = {}, a10 = {}, a11 = {};
#pragma unroll
  for (unsigned tap = 0; tap < 6u; ++tap) {
    unsigned dh = tap / 3u, dw = tap % 3u;
    // pixel w' at padded (ph = h+dh, pw = w'+dw); lane m -> w' = m / 16+m
    unsigned rb = ((b * PH_ + h + dh) * PW_ + dw) * C_;
    unsigned a0o = rb + m * C_ + kg;
    unsigned a1o = a0o + 16u * C_;
    for (unsigned kc = 0; kc < 8u; ++kc) {
      V16 A0, A1, B0, B1;
      A0.q[0] = ld4(Ain + a0o + kc * 32u);
      A0.q[1] = ld4(Ain + a0o + kc * 32u + 16u);
      A1.q[0] = ld4(Ain + a1o + kc * 32u);
      A1.q[1] = ld4(Ain + a1o + kc * 32u + 16u);
      unsigned bo = (((tap * 8u + kc) * NT + np * 2u) * 32u + lane) * 16u;
      B0.q[0] = ld4(Bp + bo);        B0.q[1] = ld4(Bp + bo + 8u);
      B1.q[0] = ld4(Bp + bo + 512u); B1.q[1] = ld4(Bp + bo + 520u);
      a00 = wmma32(A0, B0, a00);
      a01 = wmma32(A0, B1, a01);
      a10 = wmma32(A1, B0, a10);
      a11 = wmma32(A1, B1, a11);
    }
  }
  unsigned hi8 = (lane >> 4) * 8u;
  unsigned n0  = np * 32u + (lane & 15u);
  unsigned n1  = n0 + 16u;
  float bs0 = bias[n0], bs1 = bias[n1];
  unsigned Cout = NT * 16u;
#pragma unroll
  for (unsigned r = 0; r < 8u; ++r) {
    unsigned w0 = r + hi8, w1 = 16u + r + hi8;
    float y00 = a00[r] + bs0, y01 = a01[r] + bs1;
    float y10 = a10[r] + bs0, y11 = a11[r] + bs1;
    if (mode == 0) {
      unsigned p0 = ((b * PH_ + h + 1u) * PW_ + w0 + 2u) * C_;
      unsigned p1 = ((b * PH_ + h + 1u) * PW_ + w1 + 2u) * C_;
      out16[p0 + n0] = (_Float16)eluf(y00);
      out16[p0 + n1] = (_Float16)eluf(y01);
      out16[p1 + n0] = (_Float16)eluf(y10);
      out16[p1 + n1] = (_Float16)eluf(y11);
    } else {
      unsigned r0 = ((b * H_ + h) * W_ + w0) * Cout;
      unsigned r1 = ((b * H_ + h) * W_ + w1) * Cout;
      out32[r0 + n0] = y00;  out32[r0 + n1] = y01;
      out32[r1 + n0] = y10;  out32[r1 + n1] = y11;
    }
  }
}

// --- GLU residual update: x(NCHW) += a * sigmoid(g), y NHWC [B,S,2C] --------
__global__ void k_glu(const float* __restrict__ y, float* __restrict__ x) {
  unsigned i = blockIdx.x * blockDim.x + threadIdx.x;  // over B*C*S (NCHW)
  if (i >= B_ * C_ * S_) return;
  unsigned s = i & (S_ - 1u);
  unsigned c = (i >> 10) & (C_ - 1u);
  unsigned b = i >> 18;
  unsigned yo = (b * S_ + s) * (2u * C_);
  x[i] += y[yo + c] * sigm(y[yo + C_ + c]);
}

// --- epilogue scatter for pointwise GEMMs -----------------------------------
// 0: t16 = f16(elu(yv)) plain [B*S, C]
// 1/2: Q / K^T scatter into per-(b,head) WMMA operand tiles
// 3: V scatter into per-(b,head) B-operand tiles
// 4: out = elu(yv) + resid (NCHW f32)
__device__ __forceinline__ void ep_store(int mode, unsigned row, unsigned n, float yv,
                                         _Float16* __restrict__ o16,
                                         float* __restrict__ o32,
                                         const float* __restrict__ resid) {
  if (mode == 0) {
    o16[row * C_ + n] = (_Float16)eluf(yv);
  } else if (mode <= 2) {
    unsigned bb = row >> 10, s = row & 1023u;
    unsigned head = n >> 5, d = n & 31u;
    unsigned dhi = (d >> 3) & 1u;
    unsigned de  = (d & 7u) + ((d & 16u) ? 8u : 0u);
    unsigned idx = (((bb * HEADS + head) * 64u + (s >> 4)) * 32u
                    + ((s & 15u) + (dhi << 4))) * 16u + de;
    o16[idx] = (_Float16)yv;
  } else if (mode == 3) {
    unsigned bb = row >> 10, s = row & 1023u;
    unsigned head = n >> 5, d = n & 31u;
    unsigned kc = s >> 5, ks = s & 31u;
    unsigned khi = (ks >> 3) & 1u;
    unsigned ke  = (ks & 7u) + ((ks & 16u) ? 8u : 0u);
    unsigned idx = ((((bb * HEADS + head) * 32u + kc) * 2u + (d >> 4)) * 32u
                    + ((d & 15u) + (khi << 4))) * 16u + ke;
    o16[idx] = (_Float16)yv;
  } else {
    unsigned bb = row >> 10, s = row & 1023u;
    unsigned oi = (bb * C_ + n) * S_ + s;             // NCHW
    o32[oi] = eluf(yv) + resid[oi];
  }
}

// --- pointwise WMMA GEMM [8192x256]x[256x256], 2x2 register tiling ----------
// apad: A rows addressed through the padded elu buffer.
__global__ void __launch_bounds__(256)
k_gemm(const _Float16* __restrict__ A, int apad,
       const _Float16* __restrict__ Bp, const float* __restrict__ bias, int mode,
       _Float16* __restrict__ o16, float* __restrict__ o32,
       const float* __restrict__ resid) {
  unsigned wave = threadIdx.x >> 5;
  unsigned lane = threadIdx.x & 31u;
  unsigned tile = blockIdx.x * (blockDim.x >> 5) + wave;
  unsigned mt = tile >> 3;                 // 0..255 : 32 rows
  unsigned np = tile & 7u;                 // N pair : 32 cols
  unsigned m  = lane & 15u;
  unsigned kg = (lane >> 4) * 8u;
  unsigned row0 = mt * 32u + m;
  unsigned row1 = row0 + 16u;

  unsigned ao0, ao1;
  if (apad) {
    unsigned b0 = row0 >> 10, s0 = row0 & 1023u;
    unsigned b1 = row1 >> 10, s1 = row1 & 1023u;
    ao0 = ((b0 * PH_ + (s0 >> 5) + 1u) * PW_ + (s0 & 31u) + 2u) * C_ + kg;
    ao1 = ((b1 * PH_ + (s1 >> 5) + 1u) * PW_ + (s1 & 31u) + 2u) * C_ + kg;
  } else {
    ao0 = row0 * C_ + kg;
    ao1 = row1 * C_ + kg;
  }

  v8f a00 = {}, a01 = {}, a10 = {}, a11 = {};
  for (unsigned kc = 0; kc < 8u; ++kc) {
    V16 A0, A1, B0, B1;
    A0.q[0] = ld4(A + ao0 + kc * 32u);
    A0.q[1] = ld4(A + ao0 + kc * 32u + 16u);
    A1.q[0] = ld4(A + ao1 + kc * 32u);
    A1.q[1] = ld4(A + ao1 + kc * 32u + 16u);
    unsigned bo = ((kc * 16u + np * 2u) * 32u + lane) * 16u;
    B0.q[0] = ld4(Bp + bo);        B0.q[1] = ld4(Bp + bo + 8u);
    B1.q[0] = ld4(Bp + bo + 512u); B1.q[1] = ld4(Bp + bo + 520u);
    a00 = wmma32(A0, B0, a00);
    a01 = wmma32(A0, B1, a01);
    a10 = wmma32(A1, B0, a10);
    a11 = wmma32(A1, B1, a11);
  }
  unsigned hi8 = (lane >> 4) * 8u;
  unsigned n0  = np * 32u + (lane & 15u);
  unsigned n1  = n0 + 16u;
  float bs0 = bias[n0], bs1 = bias[n1];
#pragma unroll
  for (unsigned r = 0; r < 8u; ++r) {
    unsigned ra = mt * 32u + r + hi8;
    unsigned rb = ra + 16u;
    ep_store(mode, ra, n0, a00[r] + bs0, o16, o32, resid);
    ep_store(mode, ra, n1, a01[r] + bs1, o16, o32, resid);
    ep_store(mode, rb, n0, a10[r] + bs0, o16, o32, resid);
    ep_store(mode, rb, n1, a11[r] + bs1, o16, o32, resid);
  }
}

// --- flash attention: 1 wave per (b, head, 16-query tile) -------------------
__global__ void __launch_bounds__(32)
k_attn(const _Float16* __restrict__ Qp, const _Float16* __restrict__ Kp,
       const _Float16* __restrict__ Vp, _Float16* __restrict__ att) {
  __shared__ alignas(16) _Float16 P[16][32];
  unsigned lane = threadIdx.x;
  unsigned blk  = blockIdx.x;              // B*HEADS*64
  unsigned qt = blk & 63u;
  unsigned bh = blk >> 6;
  unsigned m   = lane & 15u;
  unsigned kg  = (lane >> 4) * 8u;
  unsigned hi8 = (lane >> 4) * 8u;
  unsigned n   = lane & 15u;

  V16 qa;
  unsigned qo = ((bh * 64u + qt) * 32u + lane) * 16u;
  qa.q[0] = ld4(Qp + qo);
  qa.q[1] = ld4(Qp + qo + 8u);

  v8f acc0 = {}, acc1 = {};
  float mrow[8], lrow[8];
#pragma unroll
  for (int r = 0; r < 8; ++r) { mrow[r] = NEGINF; lrow[r] = 0.f; }
  const float scale = 0.17677669529663687f;    // 1/sqrt(32)
  unsigned kcend = (qt * 16u + 15u) >> 5;

  for (unsigned kc = 0; kc <= kcend; ++kc) {
    V16 b0, b1;
    unsigned ko = ((bh * 64u + kc * 2u) * 32u + lane) * 16u;
    b0.q[0] = ld4(Kp + ko);        b0.q[1] = ld4(Kp + ko + 8u);
    b1.q[0] = ld4(Kp + ko + 512u); b1.q[1] = ld4(Kp + ko + 520u);

    v8f s0 = {}, s1 = {};
    s0 = wmma32(qa, b0, s0);
    s1 = wmma32(qa, b1, s1);

    unsigned k0 = kc * 32u + n;
    unsigned k1 = kc * 32u + 16u + n;
    float p0[8], p1[8];
#pragma unroll
    for (unsigned r = 0; r < 8u; ++r) {
      unsigned qi = qt * 16u + r + hi8;
      bool v0ok = (k0 <= qi), v1ok = (k1 <= qi);
      float a0 = v0ok ? s0[r] * scale : NEGINF;
      float a1 = v1ok ? s1[r] * scale : NEGINF;
      float mx = fmaxf(a0, a1);
      for (int off = 1; off < 16; off <<= 1) mx = fmaxf(mx, __shfl_xor(mx, off, 32));
      float mn = fmaxf(mrow[r], mx);
      float e0 = v0ok ? __expf(a0 - mn) : 0.f;
      float e1 = v1ok ? __expf(a1 - mn) : 0.f;
      float sm = e0 + e1;
      for (int off = 1; off < 16; off <<= 1) sm += __shfl_xor(sm, off, 32);
      float corr = __expf(mrow[r] - mn);
      lrow[r] = lrow[r] * corr + sm;
      acc0[r] *= corr;
      acc1[r] *= corr;
      mrow[r] = mn;
      p0[r] = e0; p1[r] = e1;
    }
#pragma unroll
    for (unsigned r = 0; r < 8u; ++r) {
      P[r + hi8][n]       = (_Float16)p0[r];
      P[r + hi8][16u + n] = (_Float16)p1[r];
    }
    __syncthreads();                       // 1-wave WG: compiler fence / S_NOP
    V16 pa;
    pa.q[0] = ld4(&P[m][kg]);
    pa.q[1] = ld4(&P[m][16u + kg]);
    __syncthreads();

    unsigned vo = ((bh * 32u + kc) * 2u * 32u + lane) * 16u;
    V16 v0, v1;
    v0.q[0] = ld4(Vp + vo);        v0.q[1] = ld4(Vp + vo + 8u);
    v1.q[0] = ld4(Vp + vo + 512u); v1.q[1] = ld4(Vp + vo + 520u);
    acc0 = wmma32(pa, v0, acc0);
    acc1 = wmma32(pa, v1, acc1);
  }

  unsigned b    = bh >> 3;
  unsigned head = bh & 7u;
#pragma unroll
  for (unsigned r = 0; r < 8u; ++r) {
    unsigned s = qt * 16u + r + hi8;
    float inv = 1.f / lrow[r];
    unsigned ao = (b * S_ + s) * C_ + head * 32u;     // c = head*32 + d
    att[ao + n]       = (_Float16)eluf(acc0[r] * inv); // includes first elu of pw_out
    att[ao + 16u + n] = (_Float16)eluf(acc1[r] * inv);
  }
}

// ---------------------------------------------------------------------------
extern "C" void kernel_launch(void* const* d_in, const int* in_sizes, int n_in,
                              void* d_out, int out_size, void* d_ws, size_t ws_size,
                              hipStream_t stream) {
  const float* input    = (const float*)d_in[0];
  const float* rb_w_in  = (const float*)d_in[1];
  const float* rb_b_in  = (const float*)d_in[2];
  const float* rb_w_out = (const float*)d_in[3];
  const float* rb_b_out = (const float*)d_in[4];
  const float* pw_in_w  = (const float*)d_in[5];
  const float* pw_in_b  = (const float*)d_in[6];
  const float* wq = (const float*)d_in[7];
  const float* bq = (const float*)d_in[8];
  const float* wk = (const float*)d_in[9];
  const float* bk = (const float*)d_in[10];
  const float* wv = (const float*)d_in[11];
  const float* bv = (const float*)d_in[12];
  const float* pw_out_w = (const float*)d_in[13];
  const float* pw_out_b = (const float*)d_in[14];

  char* ws = (char*)d_ws;
  float*     buf_x = (float*)ws;                       //  8 MiB  NCHW f32 activation
  float*     buf_y = (float*)(ws + (8u  << 20));       // 16 MiB  conv2 out NHWC; att16 reuse
  _Float16*  att16 = (_Float16*)buf_y;                 //  4 MiB  (buf_y dead after backbone)
  _Float16*  ep16  = (_Float16*)(ws + (24u << 20));    //  ~4.6 MiB padded elu NHWC
  _Float16*  ep2   = (_Float16*)(ws + (29u << 20));    //  ~4.6 MiB padded conv1 out
  _Float16*  t16   = (_Float16*)(ws + (34u << 20));    //  4 MiB  plain [B*S, C]
  _Float16*  q16   = (_Float16*)(ws + (38u << 20));
  _Float16*  k16   = (_Float16*)(ws + (42u << 20));
  _Float16*  v16   = (_Float16*)(ws + (46u << 20));
  _Float16*  wpk   = (_Float16*)(ws + (50u << 20));    // packed weights ~5.2 MiB
  _Float16*  pkC1  = wpk;                              // 2*6*65536 halves
  _Float16*  pkC2  = wpk + 786432;                     // 2*6*131072 halves
  _Float16*  pkPW  = pkC2 + 1572864;                   // 5*65536 halves

  const unsigned NELEM = B_ * C_ * S_;                 // 2,097,152
  const unsigned EB    = (NELEM + 255u) / 256u;
  const unsigned NPAD  = B_ * PH_ * PW_ * C_;          // 2,297,856
  const unsigned PB    = (NPAD + 255u) / 256u;

  // --- pack all weights into WMMA B-operand layout ---
  for (unsigned r = 0; r < 2u; ++r)
    for (unsigned tap = 0; tap < 6u; ++tap) {
      k_pack<<<16, 256, 0, stream>>>(
          rb_w_in + (size_t)r * 393216, pkC1 + (size_t)(r * 6u + tap) * 65536,
          8, 16, 1536, 6, tap);
      k_pack<<<32, 256, 0, stream>>>(
          rb_w_out + (size_t)r * 786432, pkC2 + (size_t)(r * 6u + tap) * 131072,
          8, 32, 1536, 6, tap);
    }
  k_pack<<<16, 256, 0, stream>>>(pw_in_w,  pkPW,              8, 16, 256, 1, 0);
  k_pack<<<16, 256, 0, stream>>>(wq,       pkPW + 65536,      8, 16, 256, 1, 0);
  k_pack<<<16, 256, 0, stream>>>(wk,       pkPW + 2 * 65536,  8, 16, 256, 1, 0);
  k_pack<<<16, 256, 0, stream>>>(wv,       pkPW + 3 * 65536,  8, 16, 256, 1, 0);
  k_pack<<<16, 256, 0, stream>>>(pw_out_w, pkPW + 4 * 65536,  8, 16, 256, 1, 0);

  // --- x = input; zero the conv1-output halo once per launch ---
  hipMemcpyAsync(buf_x, input, (size_t)NELEM * sizeof(float),
                 hipMemcpyDeviceToDevice, stream);
  k_fill0<<<(NPAD / 8u + 255u) / 256u, 256, 0, stream>>>((uint4*)ep2, NPAD / 8u);

  // --- backbone: 2 causal-conv GLU residual blocks ---
  for (unsigned r = 0; r < 2u; ++r) {
    k_prep_elu_pad<<<PB, 256, 0, stream>>>(buf_x, ep16);
    k_conv<<<256, 256, 0, stream>>>(ep16, pkC1 + (size_t)r * 393216,
                                    rb_b_in + r * 256, 8, ep2, nullptr, 0);
    k_conv<<<512, 256, 0, stream>>>(ep2, pkC2 + (size_t)r * 786432,
                                    rb_b_out + r * 512, 16, nullptr, buf_y, 1);
    k_glu<<<EB, 256, 0, stream>>>(buf_y, buf_x);
  }

  // --- attention branch ---
  k_prep_elu_pad<<<PB, 256, 0, stream>>>(buf_x, ep16);
  k_gemm<<<256, 256, 0, stream>>>(ep16, 1, pkPW,             pw_in_b, 0, t16, nullptr, nullptr);
  k_gemm<<<256, 256, 0, stream>>>(t16,  0, pkPW + 65536,     bq,      1, q16, nullptr, nullptr);
  k_gemm<<<256, 256, 0, stream>>>(t16,  0, pkPW + 2 * 65536, bk,      2, k16, nullptr, nullptr);
  k_gemm<<<256, 256, 0, stream>>>(t16,  0, pkPW + 3 * 65536, bv,      3, v16, nullptr, nullptr);
  k_attn<<<B_ * HEADS * 64u, 32, 0, stream>>>(q16, k16, v16, att16);
  k_gemm<<<256, 256, 0, stream>>>(att16, 0, pkPW + 4 * 65536, pw_out_b, 4,
                                  nullptr, (float*)d_out, buf_x);
  (void)in_sizes; (void)n_in; (void)out_size; (void)ws_size;
}
